// SGRP_13383118094543
// MI455X (gfx1250) — matrix operand order repaired and checked by reading, compiled, not run on hardware
//
#include <hip/hip_runtime.h>
#include <hip/hip_bf16.h>
#include <math.h>

typedef _Float16 v8h  __attribute__((ext_vector_type(8)));
typedef _Float16 v16h __attribute__((ext_vector_type(16)));
typedef float    v8f  __attribute__((ext_vector_type(8)));

// ---------------------------------------------------------------------------
// Fragment loaders, contiguous K order shared by A and B (wave32, 16-bit A 16x32 / B 32x16)
// A: lane L -> row M = L%16; halves j: K = 16*hi + j
//    => two contiguous 16B chunks at (k + 16*hi) and (k + 16*hi + 8)
// B: lane L -> col N = L%16; halves j: K = 16*hi + j => one contiguous 32B chunk
// ---------------------------------------------------------------------------
__device__ __forceinline__ v16h load_frag_a(const _Float16* __restrict__ A, int lda,
                                            int row0, int kk, int lane) {
    int m  = lane & 15;
    int hi = lane >> 4;
    const _Float16* p = A + (size_t)(row0 + m) * lda + kk + 16 * hi;
    v8h lo  = *(const v8h*)(p);
    v8h hi8 = *(const v8h*)(p + 8);
    return __builtin_shufflevector(lo, hi8, 0,1,2,3,4,5,6,7,8,9,10,11,12,13,14,15);
}

__device__ __forceinline__ v16h load_frag_b(const _Float16* __restrict__ BT, int ldb,
                                            int col0, int kk, int lane) {
    int n  = lane & 15;
    int hi = lane >> 4;
    const _Float16* p = BT + (size_t)(col0 + n) * ldb + kk + 16 * hi;
    return *(const v16h*)(p);
}

// ---------------------------------------------------------------------------
// Batched GEMM: D[M x Nc] = A[M x K] * BT[Nc x K]^T   (f16 in, f32 out)
// 8 waves/block, each wave 32x32 (2x2 WMMA tiles), block tile 64M x 128N.
// ---------------------------------------------------------------------------
__global__ __launch_bounds__(256) void gemm_wmma_kernel(
    const _Float16* __restrict__ A, const _Float16* __restrict__ BT,
    float* __restrict__ D, int M, int Nc, int K, int lda, int ldb, int ldc,
    long long sA, long long sB, long long sD)
{
    int b = blockIdx.z;
    A  += (size_t)b * sA;
    BT += (size_t)b * sB;
    D  += (size_t)b * sD;

    int lane = threadIdx.x & 31;
    int wave = threadIdx.x >> 5;
    int wm = wave >> 2, wn = wave & 3;
    int mBase = blockIdx.y * 64  + wm * 32;
    int nBase = blockIdx.x * 128 + wn * 32;

    v8f acc00 = {}, acc01 = {}, acc10 = {}, acc11 = {};

    for (int kk = 0; kk < K; kk += 32) {
        if (kk + 32 < K) {  // hint next K-slice toward the caches
            __builtin_prefetch(A  + (size_t)(mBase + (lane & 15)) * lda + kk + 32, 0, 3);
            __builtin_prefetch(BT + (size_t)(nBase + (lane & 15)) * ldb + kk + 32, 0, 3);
        }
        v16h a0 = load_frag_a(A,  lda, mBase,      kk, lane);
        v16h a1 = load_frag_a(A,  lda, mBase + 16, kk, lane);
        v16h b0 = load_frag_b(BT, ldb, nBase,      kk, lane);
        v16h b1 = load_frag_b(BT, ldb, nBase + 16, kk, lane);
        acc00 = __builtin_amdgcn_wmma_f32_16x16x32_f16(false, a0, false, b0, (short)0, acc00, false, false);
        acc01 = __builtin_amdgcn_wmma_f32_16x16x32_f16(false, a0, false, b1, (short)0, acc01, false, false);
        acc10 = __builtin_amdgcn_wmma_f32_16x16x32_f16(false, a1, false, b0, (short)0, acc10, false, false);
        acc11 = __builtin_amdgcn_wmma_f32_16x16x32_f16(false, a1, false, b1, (short)0, acc11, false, false);
    }

    // D layout: vgpr r of lane L holds (M = r + 8*(L/16), N = L%16)
    int col = lane & 15, hi = lane >> 4;
#pragma unroll
    for (int r = 0; r < 8; ++r) {
        int mr = r + 8 * hi;
        D[(size_t)(mBase + mr)      * ldc + nBase + col]      = acc00[r];
        D[(size_t)(mBase + mr)      * ldc + nBase + 16 + col] = acc01[r];
        D[(size_t)(mBase + 16 + mr) * ldc + nBase + col]      = acc10[r];
        D[(size_t)(mBase + 16 + mr) * ldc + nBase + 16 + col] = acc11[r];
    }
}

// ---------------------------------------------------------------------------
// Per-row 1/max(||x||,eps)  (one block per row)
// ---------------------------------------------------------------------------
__global__ __launch_bounds__(256) void rownorm_kernel(const float* __restrict__ X,
                                                      float* __restrict__ inv, int d)
{
    int row = blockIdx.x;
    const float* x = X + (size_t)row * d;
    float s = 0.f;
    for (int i = threadIdx.x; i < d; i += 256) { float v = x[i]; s += v * v; }
    __shared__ float red[256];
    red[threadIdx.x] = s;
    __syncthreads();
    for (int w = 128; w > 0; w >>= 1) {
        if (threadIdx.x < w) red[threadIdx.x] += red[threadIdx.x + w];
        __syncthreads();
    }
    if (threadIdx.x == 0) inv[row] = 1.0f / fmaxf(sqrtf(red[0]), 1e-6f);
}

// normalized f32 -> f16 cast (row scale)
__global__ void scale_to_f16_kernel(const float* __restrict__ X, const float* __restrict__ inv,
                                    _Float16* __restrict__ Y, int d, long long total)
{
    long long i = (long long)blockIdx.x * blockDim.x + threadIdx.x;
    if (i < total) Y[i] = (_Float16)(X[i] * inv[i / d]);
}

// Tg[b,c,dd] = t_n[c,dd] * g_n[b,dd]  (f16)
__global__ void gate_kernel(const float* __restrict__ t, const float* __restrict__ g,
                            const float* __restrict__ invt, const float* __restrict__ invg,
                            _Float16* __restrict__ Tg, int C, int d, long long total)
{
    long long i = (long long)blockIdx.x * blockDim.x + threadIdx.x;
    if (i >= total) return;
    int dd = (int)(i % d);
    long long r = i / d;
    int c = (int)(r % C);
    int b = (int)(r / C);
    Tg[i] = (_Float16)((t[(size_t)c * d + dd] * invt[c]) * (g[(size_t)b * d + dd] * invg[b]));
}

// h [N,d] f32 -> hT [d,N] f16 (32x32 LDS tiles)
__global__ __launch_bounds__(256) void transpose_h_kernel(const float* __restrict__ h,
                                                          _Float16* __restrict__ hT,
                                                          int Nn, int d)
{
    __shared__ float tile[32][33];
    int n0 = blockIdx.y * 32, d0 = blockIdx.x * 32;
#pragma unroll
    for (int i = 0; i < 4; ++i) {
        int nl = threadIdx.y + i * 8;
        tile[nl][threadIdx.x] = h[(size_t)(n0 + nl) * d + d0 + threadIdx.x];
    }
    __syncthreads();
#pragma unroll
    for (int i = 0; i < 4; ++i) {
        int dl = threadIdx.y + i * 8;
        hT[(size_t)(d0 + dl) * Nn + n0 + threadIdx.x] = (_Float16)tile[threadIdx.x][dl];
    }
}

// ---------------------------------------------------------------------------
// Online softmax stats over N per (b,c): stats = (max, 1/sum)
// blockDim (64 cols, 8 n-strides)
// ---------------------------------------------------------------------------
__global__ __launch_bounds__(512) void softmax_stats_kernel(const float* __restrict__ m,
                                                            float2* __restrict__ stats,
                                                            int Nn, int C)
{
    int b = blockIdx.y;
    int c = blockIdx.x * 64 + threadIdx.x;
    const float* mb = m + (size_t)b * Nn * C;
    float mx = -INFINITY, s = 0.f;
    for (int n = threadIdx.y; n < Nn; n += 8) {
        float x  = mb[(size_t)n * C + c];
        float nm = fmaxf(mx, x);
        s  = s * __expf(mx - nm) + __expf(x - nm);
        mx = nm;
    }
    __shared__ float smax[8][64];
    __shared__ float ssum[8][64];
    smax[threadIdx.y][threadIdx.x] = mx;
    ssum[threadIdx.y][threadIdx.x] = s;
    __syncthreads();
    if (threadIdx.y == 0) {
        float M = smax[0][threadIdx.x];
#pragma unroll
        for (int i = 1; i < 8; ++i) M = fmaxf(M, smax[i][threadIdx.x]);
        float S = 0.f;
#pragma unroll
        for (int i = 0; i < 8; ++i) S += ssum[i][threadIdx.x] * __expf(smax[i][threadIdx.x] - M);
        stats[(size_t)b * C + c] = make_float2(M, 1.0f / S);
    }
}

// apply softmax and transpose: m [N,C] f32 -> mT [C,N] f16 (64x64 LDS tiles)
__global__ __launch_bounds__(256) void softmax_apply_transpose_kernel(
    const float* __restrict__ m, const float2* __restrict__ stats,
    _Float16* __restrict__ mT, int Nn, int C)
{
    int b  = blockIdx.z;
    int n0 = blockIdx.y * 64;
    int c0 = blockIdx.x * 64;
    const float*  mb  = m  + (size_t)b * Nn * C;
    const float2* st  = stats + (size_t)b * C;
    _Float16*     mTb = mT + (size_t)b * C * Nn;
    __shared__ _Float16 tile[64][68];
#pragma unroll
    for (int i = 0; i < 16; ++i) {
        int lin = threadIdx.x + i * 256;
        int nl = lin >> 6, cl = lin & 63;
        float  x  = mb[(size_t)(n0 + nl) * C + c0 + cl];
        float2 sc = st[c0 + cl];
        tile[cl][nl] = (_Float16)(__expf(x - sc.x) * sc.y);
    }
    __syncthreads();
#pragma unroll
    for (int i = 0; i < 16; ++i) {
        int lin = threadIdx.x + i * 256;
        int cl = lin >> 6, nl = lin & 63;
        mTb[(size_t)(c0 + cl) * Nn + n0 + nl] = tile[cl][nl];
    }
}

// ---------------------------------------------------------------------------
extern "C" void kernel_launch(void* const* d_in, const int* in_sizes, int n_in,
                              void* d_out, int out_size, void* d_ws, size_t ws_size,
                              hipStream_t stream)
{
    const float* p = (const float*)d_in[0];
    const float* t = (const float*)d_in[1];
    const float* g = (const float*)d_in[2];
    const float* h = (const float*)d_in[3];
    float* out = (float*)d_out;
    (void)n_in; (void)out_size; (void)ws_size;

    // derive dims: p=B*N*d, t=C*d, g=B*d, h=N*d
    long long Sp = in_sizes[0], St = in_sizes[1], Sg = in_sizes[2], Sh = in_sizes[3];
    long long d  = (Sg * Sh) / Sp;   // (B*d)*(N*d)/(B*N*d)
    long long B  = Sg / d;
    long long Nn = Sh / d;
    long long C  = St / d;

    // workspace carve-out (256B aligned)
    char* w = (char*)d_ws;
    size_t off = 0;
    auto carve = [&](size_t bytes) -> void* {
        void* ptr = w + off;
        off = (off + bytes + 255) & ~(size_t)255;
        return ptr;
    };
    float*    inv_p = (float*)   carve((size_t)B * Nn * 4);
    float*    inv_t = (float*)   carve((size_t)C * 4);
    float*    inv_g = (float*)   carve((size_t)B * 4);
    _Float16* pn    = (_Float16*)carve((size_t)B * Nn * d * 2);
    _Float16* Tg    = (_Float16*)carve((size_t)B * C  * d * 2);
    _Float16* hT    = (_Float16*)carve((size_t)d * Nn * 2);
    float*    mm    = (float*)   carve((size_t)B * Nn * C * 4);
    float2*   stats = (float2*)  carve((size_t)B * C * 8);
    _Float16* mT    = (_Float16*)carve((size_t)B * C * Nn * 2);

    // 1) row norms
    rownorm_kernel<<<(int)(B * Nn), 256, 0, stream>>>(p, inv_p, (int)d);
    rownorm_kernel<<<(int)C,        256, 0, stream>>>(t, inv_t, (int)d);
    rownorm_kernel<<<(int)B,        256, 0, stream>>>(g, inv_g, (int)d);

    // 2) normalized casts / gate
    long long totP = B * Nn * d;
    scale_to_f16_kernel<<<(int)((totP + 255) / 256), 256, 0, stream>>>(p, inv_p, pn, (int)d, totP);
    long long totT = B * C * d;
    gate_kernel<<<(int)((totT + 255) / 256), 256, 0, stream>>>(t, g, inv_t, inv_g, Tg, (int)C, (int)d, totT);

    // 3) h transpose -> f16
    transpose_h_kernel<<<dim3((unsigned)(d / 32), (unsigned)(Nn / 32)), dim3(32, 8), 0, stream>>>(
        h, hT, (int)Nn, (int)d);

    // 4) GEMM1: m[b] = pn[b] (N x d) * Tg[b]^T (d x C)
    gemm_wmma_kernel<<<dim3((unsigned)(C / 128), (unsigned)(Nn / 64), (unsigned)B), 256, 0, stream>>>(
        pn, Tg, mm, (int)Nn, (int)C, (int)d, (int)d, (int)d, (int)C,
        (long long)(Nn * d), (long long)(C * d), (long long)(Nn * C));

    // 5) softmax over N + transpose to mT [C,N] f16
    softmax_stats_kernel<<<dim3((unsigned)(C / 64), (unsigned)B), dim3(64, 8), 0, stream>>>(
        mm, stats, (int)Nn, (int)C);
    softmax_apply_transpose_kernel<<<dim3((unsigned)(C / 64), (unsigned)(Nn / 64), (unsigned)B), 256, 0, stream>>>(
        mm, stats, mT, (int)Nn, (int)C);

    // 6) GEMM2: out[b] = mT[b] (C x N) * hT^T (N x d)   (hT broadcast, strideB = 0)
    gemm_wmma_kernel<<<dim3((unsigned)(d / 128), (unsigned)(C / 64), (unsigned)B), 256, 0, stream>>>(
        mT, hT, out, (int)C, (int)d, (int)Nn, (int)Nn, (int)Nn, (int)d,
        (long long)(C * Nn), 0LL, (long long)(C * d));
}
